// Attention_13572096655452
// MI455X (gfx1250) — matrix operand order repaired and checked by reading, not compiled, not run on hardware
//
#include <hip/hip_runtime.h>
#include <hip/hip_bf16.h>

// ---------------------------------------------------------------------------
// MI455X (gfx1250) spatio-temporal attention block.
//  - all heavy math:  v_wmma_f32_16x16x32_bf16 (wave32, f32 accumulate)
//  - all LDS staging: Tensor Data Mover (tensor_load_to_lds) where available,
//    with hardware row padding for conflict-free fragment reads and hardware
//    zero-fill of out-of-tensor pad rows.
// ---------------------------------------------------------------------------

typedef __bf16 bf16_t;
typedef bf16_t v16bf __attribute__((ext_vector_type(16)));
typedef float  v8f   __attribute__((ext_vector_type(8)));
typedef unsigned int u32x4 __attribute__((ext_vector_type(4)));
typedef int          i32x4 __attribute__((ext_vector_type(4)));
typedef int          i32x8 __attribute__((ext_vector_type(8)));

#if defined(__has_builtin)
#  if __has_builtin(__builtin_amdgcn_tensor_load_to_lds)
#    define HAVE_TDM 1
#  else
#    define HAVE_TDM 0
#  endif
#else
#  define HAVE_TDM 0
#endif
#if __has_include(<hip/amd_detail/amd_gfx1250_TDM.h>)
#  define TDM_SIX_ARG 1   // therock-10.0 headers -> 6-arg builtin
#else
#  define TDM_SIX_ARG 0   // ROCm 7.2 -> 5-arg builtin
#endif

union Frag16 { v16bf v; unsigned int u[8]; };

static __device__ __forceinline__ unsigned short f2bf(float x) {
  unsigned int u = __float_as_uint(x);
  unsigned int r = (u + 0x7FFFu + ((u >> 16) & 1u)) >> 16;  // RNE
  return (unsigned short)r;
}
static __device__ __forceinline__ float bf2f(unsigned short b) {
  return __uint_as_float(((unsigned int)b) << 16);
}
static __device__ __forceinline__ v8f wmma_bf16(const Frag16& a, const Frag16& b, v8f c) {
  return __builtin_amdgcn_wmma_f32_16x16x32_bf16(
      false, a.v, false, b.v, (short)0, c, false, false);
}

// ---------------------------------------------------------------------------
// TDM: load a [tile1 x tile0] (optionally x tile2) bf16 tile into LDS.
// Strides in elements (data_size = 2B). pad_code: DWORDs-per-pad-interval
// encode (3 => every 16 DWORDs, 4 => every 32 DWORDs); pad_amount fixed at
// 1 DWORD, giving LDS row strides of tile0 + 2 bf16 elements.
// lds_addr: low 32 bits of a flat LDS-aperture pointer == LDS byte offset.
// ---------------------------------------------------------------------------
#if HAVE_TDM
static __device__ __forceinline__ void tdm_load_tile(
    unsigned lds_addr, const void* gptr,
    unsigned td0, unsigned td1, unsigned td2,
    unsigned tile0, unsigned tile1, unsigned tile2,
    unsigned long long stride0, unsigned long long stride1,
    unsigned pad_code) {
  unsigned long long ga = (unsigned long long)gptr;
  u32x4 g0;
  g0.x = 1u;                                  // count=1 valid, user mode
  g0.y = lds_addr;                            // LDS byte address
  g0.z = (unsigned)ga;                        // global addr lo
  g0.w = (unsigned)((ga >> 32) & 0x01FFFFFFu) | (2u << 30);  // hi | type=2
  i32x8 g1;
  g1[0] = (int)((1u << 16) | (1u << 20) | (pad_code << 22)); // dsize=2B, pad_en
  g1[1] = (int)((td0 & 0xFFFFu) << 16);                      // tensor_dim0 lo
  g1[2] = (int)((td0 >> 16) | ((td1 & 0xFFFFu) << 16));      // d0 hi | d1 lo
  g1[3] = (int)((td1 >> 16) | (tile0 << 16));                // d1 hi | tile0
  g1[4] = (int)(tile1 | (tile2 << 16));                      // tile1 | tile2
  g1[5] = (int)(unsigned)(stride0 & 0xFFFFFFFFull);          // stride0[31:0]
  g1[6] = (int)(((unsigned)(stride0 >> 32) & 0xFFFFu) |
                (((unsigned)stride1 & 0xFFFFu) << 16));      // s0 hi | s1 lo
  g1[7] = (int)(unsigned)((stride1 >> 16) & 0xFFFFFFFFull);  // stride1[47:16]
  i32x4 g2; g2[0] = (int)td2; g2[1] = 0; g2[2] = 0; g2[3] = 0;
  i32x4 g3; g3[0] = 0; g3[1] = 0; g3[2] = 0; g3[3] = 0;
#if TDM_SIX_ARG
  i32x8 g4; for (int i = 0; i < 8; ++i) g4[i] = 0;
  __builtin_amdgcn_tensor_load_to_lds(g0, g1, g2, g3, g4, 0);
#else
  __builtin_amdgcn_tensor_load_to_lds(g0, g1, g2, g3, 0);
#endif
}
#endif  // HAVE_TDM

// ---------------------------------------------------------------------------
// f32 -> bf16 conversion
// ---------------------------------------------------------------------------
__global__ void cvt_f32_to_bf16(const float* __restrict__ src,
                                unsigned short* __restrict__ dst, int n) {
  for (int i = blockIdx.x * blockDim.x + threadIdx.x; i < n;
       i += gridDim.x * blockDim.x)
    dst[i] = f2bf(src[i]);
}

// ---------------------------------------------------------------------------
// GEMM: Y[M,N] = A[M,K](bf16) * W[N,K](bf16)^T + bias
//   MODE 0: Y bf16
//   MODE 1: Y f32 + Y bf16 (dual write)
//   MODE 2: Y f32 = resid + (acc + bias) * alpha
// 256 threads / 8 waves, 128x128 tile, K-step 32, TDM double buffering.
// ---------------------------------------------------------------------------
#define LDA 34                       // LDS row stride (17 DWORDs, odd)
#define GEMM_TILE (128 * LDA)        // ushorts per staged matrix
#define GEMM_LDS_BYTES (4 * GEMM_TILE * 2)

template <int MODE>
__global__ __launch_bounds__(256) void gemm_bf16_wmma(
    const unsigned short* __restrict__ Abf, const unsigned short* __restrict__ Wb,
    const float* __restrict__ bias, float* __restrict__ outF,
    unsigned short* __restrict__ outB, const float* __restrict__ resid,
    const float* __restrict__ alpha, int M, int N, int K) {
  extern __shared__ unsigned short lds[];  // [2 bufs][A|B][128*LDA]

  const int tid  = threadIdx.x;
  const int lane = tid & 31;
  const int w    = tid >> 5;
  const int h    = lane >> 4;
  const int ln   = lane & 15;
  const int wm   = w & 1;
  const int wn   = w >> 1;
  const int m0   = blockIdx.x * 128;
  const int n0   = blockIdx.y * 128;

  v8f acc[4][2];
  for (int i = 0; i < 4; ++i)
    for (int j = 0; j < 2; ++j)
      for (int e = 0; e < 8; ++e) acc[i][j][e] = 0.0f;

#if HAVE_TDM
  if (w == 0) {  // prologue: stream tile 0 into buffer 0
    tdm_load_tile((unsigned)(unsigned long long)&lds[0],
                  Abf + (size_t)m0 * K, 32, 128, 0, 32, 128, 0,
                  (unsigned long long)K, 0, 3);
    tdm_load_tile((unsigned)(unsigned long long)&lds[GEMM_TILE],
                  Wb + (size_t)n0 * K, 32, 128, 0, 32, 128, 0,
                  (unsigned long long)K, 0, 3);
  }
#endif

  for (int ks = 0; ks < K; ks += 32) {
    const int cur = (ks >> 5) & 1;
    unsigned short* As = &lds[cur * 2 * GEMM_TILE];
    unsigned short* Bs = As + GEMM_TILE;
#if HAVE_TDM
    if (w == 0) {
      if (ks + 32 < K) {  // prefetch next tile into the other buffer
        unsigned short* An = &lds[(cur ^ 1) * 2 * GEMM_TILE];
        tdm_load_tile((unsigned)(unsigned long long)An,
                      Abf + (size_t)m0 * K + ks + 32, 32, 128, 0, 32, 128, 0,
                      (unsigned long long)K, 0, 3);
        tdm_load_tile((unsigned)(unsigned long long)(An + GEMM_TILE),
                      Wb + (size_t)n0 * K + ks + 32, 32, 128, 0, 32, 128, 0,
                      (unsigned long long)K, 0, 3);
        __builtin_amdgcn_s_wait_tensorcnt(2);  // current tile's 2 DMAs done
      } else {
        __builtin_amdgcn_s_wait_tensorcnt(0);
      }
    }
    __syncthreads();  // data visible to all waves
#else
    // fallback: cooperative raw bf16 copies
    for (int q = 0; q < 2; ++q) {
      int idx = tid + 256 * q;                 // 8-bf16 chunk
      int r = idx >> 2, kc = (idx & 3) * 8;
      const uint4 av = *(const uint4*)(Abf + (size_t)(m0 + r) * K + ks + kc);
      unsigned int* p = (unsigned int*)&As[r * LDA + kc];
      p[0] = av.x; p[1] = av.y; p[2] = av.z; p[3] = av.w;
      const uint4 bv = *(const uint4*)(Wb + (size_t)(n0 + r) * K + ks + kc);
      unsigned int* pb = (unsigned int*)&Bs[r * LDA + kc];
      pb[0] = bv.x; pb[1] = bv.y; pb[2] = bv.z; pb[3] = bv.w;
    }
    __syncthreads();
#endif

    Frag16 a[4], b[2];
    for (int fm = 0; fm < 4; ++fm) {
      int r = wm * 64 + fm * 16 + ln;
      for (int v = 0; v < 8; ++v) {
        int kk = h * 16 + 2 * v;                         // contiguous k = 16h + s (matches B)
        a[fm].u[v] = *(const unsigned int*)&As[r * LDA + kk];
      }
    }
    for (int fn = 0; fn < 2; ++fn) {
      int c = wn * 32 + fn * 16 + ln;
      for (int v = 0; v < 8; ++v) {
        int kk = h * 16 + 2 * v;                         // contiguous k = 16h + s (matches A)
        b[fn].u[v] = *(const unsigned int*)&Bs[c * LDA + kk];
      }
    }
    for (int fm = 0; fm < 4; ++fm)
      for (int fn = 0; fn < 2; ++fn)
        acc[fm][fn] = wmma_bf16(a[fm], b[fn], acc[fm][fn]);
    __syncthreads();  // finished reading buffer before TDM overwrites it
  }

  // epilogue (one straight-line variant per template instantiation)
  for (int fm = 0; fm < 4; ++fm) {
    int rowb = m0 + wm * 64 + fm * 16 + 8 * h;
    for (int fn = 0; fn < 2; ++fn) {
      int col = n0 + wn * 32 + fn * 16 + ln;
      float bv = bias[col];
      for (int v = 0; v < 8; ++v) {
        float  val = acc[fm][fn][v] + bv;
        size_t o   = (size_t)(rowb + v) * N + col;
        if (MODE == 0) {
          outB[o] = f2bf(val);
        } else if (MODE == 1) {
          outF[o] = val;
          outB[o] = f2bf(val);
        } else {
          outF[o] = resid[o] + val * alpha[col];
        }
      }
    }
  }
}

// ---------------------------------------------------------------------------
// Flash attention, one workgroup = (sequence, head). qkv bf16 [25088, 2304],
// channel layout [3][12][64]. K/V tiles streamed by TDM:
//   MODE 0 spatial: 2-D tile, rows seq*196+pos; hardware zero-fills rows>=196
//   MODE 1 th:      3-D tile (x=ch, y=h stride 14*2304, z=t stride 196*2304)
//   MODE 2 tw:      3-D tile (x=ch, y=w stride 2304,    z=t stride 196*2304)
// Each wave owns 16 query rows; 32 keys/step online softmax; S=QK^T and
// O+=PV both on WMMA. Output bf16 (feeds the next GEMM); tw accumulates.
// ---------------------------------------------------------------------------
#define LPAD 224
#define KLD  66   // TDM pad: 32-DWORD rows + 1 DWORD pad = 66 ushorts
#define QLD  66
#define ATTN_LDS_BYTES ((2 * LPAD * KLD + 8 * 16 * QLD) * 2)

template <int MODE>
static __device__ __forceinline__ int attn_row(int seq, int pos) {
  if (MODE == 0) return seq * 196 + pos;
  int b = seq / 14, inner = seq % 14;
  int t = pos / 14, j = pos % 14;
  if (MODE == 1) return b * 3136 + t * 196 + j * 14 + inner;  // th
  return b * 3136 + t * 196 + inner * 14 + j;                 // tw
}

template <int MODE, bool ACC>
__global__ __launch_bounds__(256) void flash_attn_wmma(
    const unsigned short* __restrict__ qkv, unsigned short* __restrict__ out,
    int L) {
  extern __shared__ unsigned short smem[];
  unsigned short* Ks = smem;                   // [224][KLD]
  unsigned short* Vs = smem + LPAD * KLD;      // [224][KLD] (row-major)
  unsigned short* Ws = smem + 2 * LPAD * KLD;  // 8 * [16][QLD] wave scratch

  const int head = blockIdx.x % 12;
  const int seq  = blockIdx.x / 12;
  const int tid  = threadIdx.x;
  const int lane = tid & 31;
  const int w    = tid >> 5;
  const int h    = lane >> 4;
  const int ln   = lane & 15;
  const int ldq  = 2304;
  const int cb   = head * 64;

#if HAVE_TDM
  if (w == 0) {
    unsigned lK = (unsigned)(unsigned long long)Ks;
    unsigned lV = (unsigned)(unsigned long long)Vs;
    if (MODE == 0) {
      size_t base = (size_t)seq * 196 * ldq + cb;
      // tensor_dim1 = 196 < tile_dim1 = 224 -> pad rows hardware-zeroed
      tdm_load_tile(lK, qkv + base + 768, 64, 196, 0, 64, 224, 0, 2304ull, 0, 4);
      tdm_load_tile(lV, qkv + base + 1536, 64, 196, 0, 64, 224, 0, 2304ull, 0, 4);
    } else {
      int b = seq / 14, inner = seq % 14;
      size_t baseRow = (MODE == 1) ? ((size_t)b * 3136 + inner)
                                   : ((size_t)b * 3136 + (size_t)inner * 14);
      unsigned long long s0 = (MODE == 1) ? (14ull * 2304) : 2304ull;
      unsigned long long s1 = 196ull * 2304;
      size_t base = baseRow * ldq + cb;
      tdm_load_tile(lK, qkv + base + 768, 64, 14, 16, 64, 14, 16, s0, s1, 4);
      tdm_load_tile(lV, qkv + base + 1536, 64, 14, 16, 64, 14, 16, s0, s1, 4);
    }
    __builtin_amdgcn_s_wait_tensorcnt(0);
  }
  __syncthreads();
#else
  for (int i = tid; i < LPAD * 64; i += 256) {
    int p = i >> 6, d = i & 63;
    unsigned short kv = 0, vv = 0;
    if (p < L) {
      size_t r = (size_t)attn_row<MODE>(seq, p) * ldq;
      kv = qkv[r + 768 + cb + d];
      vv = qkv[r + 1536 + cb + d];
    }
    Ks[p * KLD + d] = kv;
    Vs[p * KLD + d] = vv;
  }
  __syncthreads();
#endif

  unsigned short* Qs = Ws + w * 16 * QLD;  // wave-private

  for (int qb = w; qb < LPAD / 16; qb += 8) {
    // stage this wave's 16x64 Q block (raw bf16; scale applied to S later)
    for (int i = lane; i < 16 * 64; i += 32) {
      int r = i >> 6, d = i & 63;
      int p = qb * 16 + r;
      unsigned short qv = 0;
      if (p < L) {
        size_t ro = (size_t)attn_row<MODE>(seq, p) * ldq;
        qv = qkv[ro + cb + d];
      }
      Qs[r * QLD + d] = qv;  // same-wave LDS is in-order
    }
    Frag16 aq[2];
    for (int f = 0; f < 2; ++f)
      for (int v = 0; v < 8; ++v) {
        int kk = f * 32 + h * 16 + 2 * v;
        unsigned int lo = Qs[ln * QLD + kk];
        unsigned int hi = Qs[ln * QLD + kk + 1];
        aq[f].u[v] = lo | (hi << 16);
      }

    v8f o[4];
    for (int t = 0; t < 4; ++t)
      for (int e = 0; e < 8; ++e) o[t][e] = 0.0f;
    float mrow[8], lrow[8];
    for (int v = 0; v < 8; ++v) { mrow[v] = -1e30f; lrow[v] = 0.0f; }

    for (int jb = 0; jb < LPAD; jb += 32) {
      v8f s0, s1;
      for (int e = 0; e < 8; ++e) { s0[e] = 0.0f; s1[e] = 0.0f; }
      Frag16 bk;
      for (int half = 0; half < 2; ++half) {
        for (int v = 0; v < 8; ++v)
          bk.u[v] = *(const unsigned int*)&Ks[(jb + ln) * KLD + half * 32 + h * 16 + 2 * v];
        s0 = wmma_bf16(aq[half], bk, s0);
      }
      for (int half = 0; half < 2; ++half) {
        for (int v = 0; v < 8; ++v)
          bk.u[v] = *(const unsigned int*)&Ks[(jb + 16 + ln) * KLD + half * 32 + h * 16 + 2 * v];
        s1 = wmma_bf16(aq[half], bk, s1);
      }
      // scale (hd^-0.5 = 0.125) and mask padded keys
      bool pad0 = (jb + ln) >= L;
      bool pad1 = (jb + 16 + ln) >= L;
      for (int v = 0; v < 8; ++v) {
        s0[v] = pad0 ? -1e30f : s0[v] * 0.125f;
        s1[v] = pad1 ? -1e30f : s1[v] * 0.125f;
      }
      // online softmax; rows live across 16 lanes of a half-wave
      for (int v = 0; v < 8; ++v) {
        float mx = fmaxf(s0[v], s1[v]);
        for (int off = 1; off < 16; off <<= 1)
          mx = fmaxf(mx, __shfl_xor(mx, off, 16));
        float mnew = fmaxf(mrow[v], mx);
        float corr = __expf(mrow[v] - mnew);
        float p0 = __expf(s0[v] - mnew);
        float p1 = __expf(s1[v] - mnew);
        s0[v] = p0; s1[v] = p1;
        float rs = p0 + p1;
        for (int off = 1; off < 16; off <<= 1) rs += __shfl_xor(rs, off, 16);
        mrow[v] = mnew;
        lrow[v] = lrow[v] * corr + rs;
        for (int t = 0; t < 4; ++t) o[t][v] *= corr;
      }
      // re-fragment P through wave-private LDS (C layout -> A layout)
      for (int v = 0; v < 8; ++v) {
        Qs[(v + 8 * h) * QLD + ln]      = f2bf(s0[v]);
        Qs[(v + 8 * h) * QLD + 16 + ln] = f2bf(s1[v]);
      }
      Frag16 ap;
      for (int v = 0; v < 8; ++v) {
        int kk = h * 16 + 2 * v;
        unsigned int lo = Qs[ln * QLD + kk];
        unsigned int hi = Qs[ln * QLD + kk + 1];
        ap.u[v] = lo | (hi << 16);
      }
      // O += P V  (V row-major: pack pairs from adjacent key rows)
      for (int t = 0; t < 4; ++t) {
        Frag16 bv;
        for (int v = 0; v < 8; ++v) {
          int key = jb + h * 16 + 2 * v;
          unsigned int lo = Vs[key * KLD + t * 16 + ln];
          unsigned int hi = Vs[(key + 1) * KLD + t * 16 + ln];
          bv.u[v] = lo | (hi << 16);
        }
        o[t] = wmma_bf16(ap, bv, o[t]);
      }
    }

    // normalize and scatter-store bf16
    for (int v = 0; v < 8; ++v) {
      int p = qb * 16 + v + 8 * h;
      if (p < L) {
        float  inv = 1.0f / lrow[v];
        size_t ro  = (size_t)attn_row<MODE>(seq, p) * 768 + cb;
        for (int t = 0; t < 4; ++t) {
          float  val = o[t][v] * inv;
          size_t idx = ro + t * 16 + ln;
          if (ACC) out[idx] = f2bf(bf2f(out[idx]) + val);
          else     out[idx] = f2bf(val);
        }
      }
    }
  }
}

// ---------------------------------------------------------------------------
// Host orchestration
// ---------------------------------------------------------------------------
extern "C" void kernel_launch(void* const* d_in, const int* in_sizes, int n_in,
                              void* d_out, int out_size, void* d_ws,
                              size_t ws_size, hipStream_t stream) {
  (void)in_sizes; (void)n_in; (void)out_size; (void)ws_size;
  const float* x       = (const float*)d_in[0];
  const float* W_in    = (const float*)d_in[1];
  const float* b_in    = (const float*)d_in[2];
  const float* W_out   = (const float*)d_in[3];
  const float* b_out   = (const float*)d_in[4];
  const float* W_in_t  = (const float*)d_in[5];
  const float* b_in_t  = (const float*)d_in[6];
  const float* W_out_t = (const float*)d_in[7];
  const float* b_out_t = (const float*)d_in[8];
  const float* alpha   = (const float*)d_in[9];
  // d_in[10..12] = T,H,W (hardcoded 16,14,14)

  const int M = 25088, C = 768, C3 = 2304;

  char*  ws  = (char*)d_ws;
  size_t off = 0;
  auto alloc = [&](size_t bytes) {
    void* p = ws + off;
    off += (bytes + 255) & ~(size_t)255;
    return p;
  };
  unsigned short* x_bf     = (unsigned short*)alloc((size_t)M * C * 2);
  unsigned short* qkv_bf   = (unsigned short*)alloc((size_t)M * C3 * 2);
  unsigned short* Wb_in    = (unsigned short*)alloc((size_t)C3 * C * 2);
  unsigned short* Wb_out   = (unsigned short*)alloc((size_t)C * C * 2);
  unsigned short* Wb_in_t  = (unsigned short*)alloc((size_t)C3 * C * 2);
  unsigned short* Wb_out_t = (unsigned short*)alloc((size_t)C * C * 2);
  unsigned short* attn_bf  = (unsigned short*)alloc((size_t)M * C * 2);
  unsigned short* x2_bf    = (unsigned short*)alloc((size_t)M * C * 2);
  unsigned short* thtw_bf  = (unsigned short*)alloc((size_t)M * C * 2);
  float*          x2_f     = (float*)alloc((size_t)M * C * 4);

  cvt_f32_to_bf16<<<2048, 256, 0, stream>>>(x, x_bf, M * C);
  cvt_f32_to_bf16<<<1024, 256, 0, stream>>>(W_in, Wb_in, C3 * C);
  cvt_f32_to_bf16<<<512,  256, 0, stream>>>(W_out, Wb_out, C * C);
  cvt_f32_to_bf16<<<1024, 256, 0, stream>>>(W_in_t, Wb_in_t, C3 * C);
  cvt_f32_to_bf16<<<512,  256, 0, stream>>>(W_out_t, Wb_out_t, C * C);

  dim3 g_in(M / 128, C3 / 128);   // 196 x 18
  dim3 g_out(M / 128, C / 128);   // 196 x 6

  // qkv = x @ W_in^T + b_in   (bf16)
  gemm_bf16_wmma<0><<<g_in, 256, GEMM_LDS_BYTES, stream>>>(
      x_bf, Wb_in, b_in, nullptr, qkv_bf, nullptr, nullptr, M, C3, C);
  // spatial attention (128 sequences of length 196)
  flash_attn_wmma<0, false><<<128 * 12, 256, ATTN_LDS_BYTES, stream>>>(
      qkv_bf, attn_bf, 196);
  // x2 = attn @ W_out^T + b_out   (f32 for residual + bf16 for next GEMM)
  gemm_bf16_wmma<1><<<g_out, 256, GEMM_LDS_BYTES, stream>>>(
      attn_bf, Wb_out, b_out, x2_f, x2_bf, nullptr, nullptr, M, C, C);
  // qkv_t = x2 @ W_in_t^T + b_in_t   (bf16, reuse qkv buffer)
  gemm_bf16_wmma<0><<<g_in, 256, GEMM_LDS_BYTES, stream>>>(
      x2_bf, Wb_in_t, b_in_t, nullptr, qkv_bf, nullptr, nullptr, M, C3, C);
  // th (write) + tw (accumulate): 112 sequences of length 224
  flash_attn_wmma<1, false><<<112 * 12, 256, ATTN_LDS_BYTES, stream>>>(
      qkv_bf, thtw_bf, 224);
  flash_attn_wmma<2, true><<<112 * 12, 256, ATTN_LDS_BYTES, stream>>>(
      qkv_bf, thtw_bf, 224);
  // out = x2 + ((th+tw) @ W_out_t^T + b_out_t) * alpha
  gemm_bf16_wmma<2><<<g_out, 256, GEMM_LDS_BYTES, stream>>>(
      thtw_bf, Wb_out_t, b_out_t, (float*)d_out, nullptr, x2_f, alpha, M, C, C);
}
